// GATMissingEmbedder_43095701848696
// MI455X (gfx1250) — matrix operand order repaired and checked by reading, not compiled, not run on hardware
//
#include <hip/hip_runtime.h>

#define NN 10000      // nodes
#define NE 320000     // edges

typedef __attribute__((ext_vector_type(16))) __bf16          v16bf;
typedef __attribute__((ext_vector_type(8)))  float           v8f;
typedef __attribute__((ext_vector_type(8)))  unsigned short  ush8;

union Frag16 { v16bf v; ush8 u[2]; };

__device__ __forceinline__ unsigned short f2bf(float x) {
  unsigned b = __float_as_uint(x);
  b += 0x7FFFu + ((b >> 16) & 1u);          // round-to-nearest-even
  return (unsigned short)(b >> 16);
}

// order-preserving float<->uint encoding for atomic max on floats
__device__ __forceinline__ unsigned encOrd(float x) {
  unsigned b = __float_as_uint(x);
  return b ^ ((unsigned)((int)b >> 31) | 0x80000000u);
}
__device__ __forceinline__ float decOrd(unsigned u) {
  unsigned mask = ((int)u < 0) ? 0x80000000u : 0xFFFFFFFFu;
  return __uint_as_float(u ^ mask);
}

__device__ __forceinline__ float elu1(float v) {
  return v > 0.0f ? v : (__expf(v) - 1.0f);
}

// ---------------- projection: h = elu(x * proj_w + proj_b), store bf16 -------
__global__ void gat_proj_kernel(const float* __restrict__ x,
                                const float* __restrict__ pw,
                                const float* __restrict__ pb,
                                unsigned short* __restrict__ hb) {
  int i = blockIdx.x * blockDim.x + threadIdx.x;
  if (i >= NN * 128) return;
  int n = i >> 7, c = i & 127;
  hb[i] = f2bf(elu1(x[n] * pw[c] + pb[c]));
}

// ---------------- pack f32 weights [K,Nc] into WMMA B-fragment layout --------
// Bp[((kt*ntiles+nt)*32 + lane)*16 + e] = bf16(W[(kt*32 + kl)*Nc + nt*16 + nl])
//   lane<16: nl=lane,    kl=(e<8 ? e : 8+e)        lane>=16: nl=lane-16, kl=(e<8 ? 8+e : 16+e)
__global__ void gat_packw_kernel(const float* __restrict__ W,
                                 unsigned short* __restrict__ Bp,
                                 int K, int Nc) {
  int idx = blockIdx.x * blockDim.x + threadIdx.x;
  int total = K * Nc;
  if (idx >= total) return;
  int e    = idx & 15;
  int lane = (idx >> 4) & 31;
  int tile = idx >> 9;
  int ntiles = Nc >> 4;
  int nt = tile % ntiles, kt = tile / ntiles;
  int nl = lane & 15;
  int kl = ((lane >> 4) << 3) + (e & 7) + ((e >> 3) << 4);
  Bp[idx] = f2bf(W[(kt * 32 + kl) * Nc + nt * 16 + nl]);
}

// ---------------- GEMM: C[M,Nc] = A[M,K](bf16) @ Bp(packed bf16), f32 acc ----
// one wave per 16x64 output strip; grid = (M/16, Nc/64), block = 32
__global__ void gat_gemm_wmma_kernel(const unsigned short* __restrict__ A,
                                     const unsigned short* __restrict__ Bp,
                                     float* __restrict__ C,
                                     int K, int Nc) {
  const int lane   = threadIdx.x;
  const int tm     = blockIdx.x;
  const int tnBase = blockIdx.y * 4;
  const int ntiles = Nc >> 4;
  const int ktiles = K >> 5;
  const int r      = lane & 15;
  const int koff   = (lane >> 4) << 3;    // 0 or 8

  v8f acc[4];
  acc[0] = v8f{}; acc[1] = v8f{}; acc[2] = v8f{}; acc[3] = v8f{};

  const unsigned short* arow = A + (size_t)(tm * 16 + r) * K;
  for (int kt = 0; kt < ktiles; ++kt) {
    Frag16 a;
    a.u[0] = *(const ush8*)(arow + kt * 32 + koff);
    a.u[1] = *(const ush8*)(arow + kt * 32 + 16 + koff);
    const unsigned short* bbase =
        Bp + ((size_t)((kt * ntiles + tnBase) * 32 + lane) << 4);
#pragma unroll
    for (int j = 0; j < 4; ++j) {
      Frag16 b;
      b.u[0] = ((const ush8*)(bbase + j * 512))[0];
      b.u[1] = ((const ush8*)(bbase + j * 512))[1];
      acc[j] = __builtin_amdgcn_wmma_f32_16x16x32_bf16(
          false, a.v, false, b.v, (short)0, acc[j], false, false);
    }
  }

  const int rowBase = tm * 16 + ((lane >> 4) << 3);
#pragma unroll
  for (int j = 0; j < 4; ++j) {
    float* cp = C + (size_t)rowBase * Nc + (tnBase + j) * 16 + r;
#pragma unroll
    for (int v = 0; v < 8; ++v) cp[(size_t)v * Nc] = acc[j][v];
  }
}

// ---------------- attention coefficients per (node, head) -------------------
__global__ void gat_attn_kernel(const float* __restrict__ hmat,
                                const float* __restrict__ a_src,
                                const float* __restrict__ a_dst,
                                float* __restrict__ as, float* __restrict__ ad,
                                int H, int C, int F, int hshift) {
  int i = blockIdx.x * blockDim.x + threadIdx.x;
  if (i >= NN * H) return;
  int n = i >> hshift, h = i & (H - 1);
  const float* hp = hmat + (size_t)n * F + h * C;
  float s = 0.f, d = 0.f;
  for (int c = 0; c < C; ++c) {
    float v = hp[c];
    s += v * a_src[h * C + c];
    d += v * a_dst[h * C + c];
  }
  as[i] = s;
  ad[i] = d;
}

// ---------------- edge pass 1: leaky-relu logits + segment max --------------
__global__ void gat_edge_max_kernel(const int* __restrict__ srcIdx,
                                    const int* __restrict__ dstIdx,
                                    const float* __restrict__ as,
                                    const float* __restrict__ ad,
                                    float* __restrict__ ebuf,
                                    unsigned* __restrict__ emax,
                                    int H, int hshift) {
  int i = blockIdx.x * blockDim.x + threadIdx.x;
  if (i >= NE * H) return;
  int e = i >> hshift, h = i & (H - 1);
  int s = srcIdx[e], d = dstIdx[e];
  float el = as[s * H + h] + ad[d * H + h];
  el = el > 0.f ? el : 0.2f * el;
  ebuf[i] = el;
  atomicMax(&emax[d * H + h], encOrd(el));
}

// ---------------- edge pass 2: exp(el - max) + segment sum ------------------
__global__ void gat_edge_exp_kernel(const int* __restrict__ dstIdx,
                                    float* __restrict__ ebuf,
                                    const unsigned* __restrict__ emax,
                                    float* __restrict__ esum,
                                    int H, int hshift) {
  int i = blockIdx.x * blockDim.x + threadIdx.x;
  if (i >= NE * H) return;
  int e = i >> hshift, h = i & (H - 1);
  int d = dstIdx[e];
  float v = __expf(ebuf[i] - decOrd(emax[d * H + h]));
  ebuf[i] = v;
  atomicAdd(&esum[d * H + h], v);
}

// ---------------- edge pass 3: weighted scatter-add (one wave per edge) -----
__global__ void gat_edge_aggr_kernel(const int* __restrict__ srcIdx,
                                     const int* __restrict__ dstIdx,
                                     const float* __restrict__ hmat,
                                     const float* __restrict__ ebuf,
                                     const float* __restrict__ esum,
                                     float* __restrict__ aggr,
                                     int H, int F, int cshift) {
  int e    = blockIdx.x * (blockDim.x >> 5) + (threadIdx.x >> 5);
  int lane = threadIdx.x & 31;
  if (e >= NE) return;
  int s = srcIdx[e], d = dstIdx[e];
  int iters = F >> 5;
  for (int i = 0; i < iters; ++i) {
    int f = (i << 5) + lane;
    int h = f >> cshift;
    float alpha = ebuf[(size_t)e * H + h] / (esum[d * H + h] + 1e-16f);
    float v = hmat[(size_t)s * F + f] * alpha;
    atomicAdd(&aggr[(size_t)d * F + f], v);
  }
}

// ---------------- finalize layers 1/2: bias + elu -> bf16 -------------------
__global__ void gat_fin_elu_kernel(const float* __restrict__ aggr,
                                   const float* __restrict__ bias,
                                   unsigned short* __restrict__ hb,
                                   int total, int fmask) {
  int i = blockIdx.x * blockDim.x + threadIdx.x;
  if (i >= total) return;
  hb[i] = f2bf(elu1(aggr[i] + bias[i & fmask]));
}

// ---------------- finalize layer 3: bias -> f32 output ----------------------
__global__ void gat_fin_out_kernel(const float* __restrict__ aggr,
                                   const float* __restrict__ bias,
                                   float* __restrict__ out,
                                   int total, int fmask) {
  int i = blockIdx.x * blockDim.x + threadIdx.x;
  if (i >= total) return;
  out[i] = aggr[i] + bias[i & fmask];
}

// ============================ host driver ===================================
static char* wsoff(void* ws, size_t& off, size_t bytes) {
  char* p = (char*)ws + off;
  off += (bytes + 255) & ~(size_t)255;
  return p;
}

static void gat_layer(hipStream_t stream, int K, int H, int C,
                      const unsigned short* hb_in, const unsigned short* wp,
                      const float* a_src, const float* a_dst, const float* bias,
                      const int* srcIdx, const int* dstIdx,
                      float* hmat, float* aggr, float* as, float* ad,
                      unsigned* emax, float* esum, float* ebuf,
                      unsigned short* hb_out, float* out_final) {
  const int F = H * C;
  const int hshift = (H == 4) ? 2 : 0;
  const int cshift = (C == 64) ? 6 : 7;

  // GEMM: hmat[N,F] = hb_in[N,K] @ W
  dim3 gg(NN / 16, F / 64);
  gat_gemm_wmma_kernel<<<gg, 32, 0, stream>>>(hb_in, wp, hmat, K, F);

  hipMemsetAsync(emax, 0, (size_t)NN * H * 4, stream);
  hipMemsetAsync(esum, 0, (size_t)NN * H * 4, stream);
  hipMemsetAsync(aggr, 0, (size_t)NN * F * 4, stream);

  gat_attn_kernel<<<(NN * H + 255) / 256, 256, 0, stream>>>(
      hmat, a_src, a_dst, as, ad, H, C, F, hshift);

  int eth = NE * H;
  gat_edge_max_kernel<<<(eth + 255) / 256, 256, 0, stream>>>(
      srcIdx, dstIdx, as, ad, ebuf, emax, H, hshift);
  gat_edge_exp_kernel<<<(eth + 255) / 256, 256, 0, stream>>>(
      dstIdx, ebuf, emax, esum, H, hshift);
  gat_edge_aggr_kernel<<<(NE + 7) / 8, 256, 0, stream>>>(
      srcIdx, dstIdx, hmat, ebuf, esum, aggr, H, F, cshift);

  int total = NN * F;
  if (hb_out)
    gat_fin_elu_kernel<<<(total + 255) / 256, 256, 0, stream>>>(
        aggr, bias, hb_out, total, F - 1);
  else
    gat_fin_out_kernel<<<(total + 255) / 256, 256, 0, stream>>>(
        aggr, bias, out_final, total, F - 1);
}

extern "C" void kernel_launch(void* const* d_in, const int* in_sizes, int n_in,
                              void* d_out, int out_size, void* d_ws, size_t ws_size,
                              hipStream_t stream) {
  const float* x       = (const float*)d_in[0];
  const int*   srcIdx  = (const int*)d_in[1];
  const int*   dstIdx  = srcIdx + NE;
  const float* proj_w  = (const float*)d_in[2];
  const float* proj_b  = (const float*)d_in[3];
  const float* w1      = (const float*)d_in[4];
  const float* a_src1  = (const float*)d_in[5];
  const float* a_dst1  = (const float*)d_in[6];
  const float* b1      = (const float*)d_in[7];
  const float* w2      = (const float*)d_in[8];
  const float* a_src2  = (const float*)d_in[9];
  const float* a_dst2  = (const float*)d_in[10];
  const float* b2      = (const float*)d_in[11];
  const float* w3      = (const float*)d_in[12];
  const float* a_src3  = (const float*)d_in[13];
  const float* a_dst3  = (const float*)d_in[14];
  const float* b3      = (const float*)d_in[15];
  float* out           = (float*)d_out;

  size_t off = 0;
  unsigned short* hb   = (unsigned short*)wsoff(d_ws, off, (size_t)NN * 256 * 2);
  float* hmat          = (float*)wsoff(d_ws, off, (size_t)NN * 256 * 4);
  float* aggr          = (float*)wsoff(d_ws, off, (size_t)NN * 256 * 4);
  float* as            = (float*)wsoff(d_ws, off, (size_t)NN * 4 * 4);
  float* ad            = (float*)wsoff(d_ws, off, (size_t)NN * 4 * 4);
  unsigned* emax       = (unsigned*)wsoff(d_ws, off, (size_t)NN * 4 * 4);
  float* esum          = (float*)wsoff(d_ws, off, (size_t)NN * 4 * 4);
  float* ebuf          = (float*)wsoff(d_ws, off, (size_t)NE * 4 * 4);
  unsigned short* wp1  = (unsigned short*)wsoff(d_ws, off, (size_t)128 * 256 * 2);
  unsigned short* wp2  = (unsigned short*)wsoff(d_ws, off, (size_t)256 * 256 * 2);
  unsigned short* wp3  = (unsigned short*)wsoff(d_ws, off, (size_t)256 * 128 * 2);

  // pack all weights into WMMA B-fragment bf16 layout
  gat_packw_kernel<<<(128 * 256 + 255) / 256, 256, 0, stream>>>(w1, wp1, 128, 256);
  gat_packw_kernel<<<(256 * 256 + 255) / 256, 256, 0, stream>>>(w2, wp2, 256, 256);
  gat_packw_kernel<<<(256 * 128 + 255) / 256, 256, 0, stream>>>(w3, wp3, 256, 128);

  // h0 = elu(x @ proj_w + proj_b)  -> bf16 [N,128]
  gat_proj_kernel<<<(NN * 128 + 255) / 256, 256, 0, stream>>>(x, proj_w, proj_b, hb);

  // layer 1: K=128, H=4, C=64, concat -> elu -> bf16 [N,256] (in place into hb)
  gat_layer(stream, 128, 4, 64, hb, wp1, a_src1, a_dst1, b1, srcIdx, dstIdx,
            hmat, aggr, as, ad, emax, esum, ebuf, hb, nullptr);
  // layer 2: K=256, H=4, C=64, concat -> elu -> bf16 [N,256]
  gat_layer(stream, 256, 4, 64, hb, wp2, a_src2, a_dst2, b2, srcIdx, dstIdx,
            hmat, aggr, as, ad, emax, esum, ebuf, hb, nullptr);
  // layer 3: K=256, H=1, C=128, mean(=identity) + bias -> f32 output [N,128]
  gat_layer(stream, 256, 1, 128, hb, wp3, a_src3, a_dst3, b3, srcIdx, dstIdx,
            hmat, aggr, as, ad, emax, esum, ebuf, nullptr, out);
}
